// Knot_43069932044493
// MI455X (gfx1250) — matrix operand order repaired and checked by reading, not compiled, not run on hardware
//
#include <hip/hip_runtime.h>

// Problem constants from the reference: B=8, S=131072, K=16, D=8.
#define B_SZ 8
#define S_SZ 131072
#define K_SZ 16
#define D_SZ 8

typedef __attribute__((ext_vector_type(16))) _Float16 v16h;
typedef __attribute__((ext_vector_type(8)))  float    v8f;
typedef __attribute__((ext_vector_type(2)))  __fp16   h2v;  // matches cvt_pkrtz

// ---------------------------------------------------------------------------
// Workspace layout (floats):
//   ws[   0..127 ] : F' in B-fragment lane order: idx=((g*4+c)*8+v)*2+m
//                    -> F'[d=2c+g][k'=2v+m] / (2pi)  (cumulative triu sum)
//   ws[ 128..255 ] : P' same order
//   ws[ 256..1279] : A fragments, half2-packed dwords: idx=(c*8+v)*32+L
//                    A[M=L%16, K(v,g=L/16)] = reg[d,k']*(k'==M)
//   ws[1280..1295] : knotRadii[k]  (C-matrix initializer)
// 1/(2pi) folded into F'/P' so v_cos_f32 (cos(2pi*x)) is fed directly.
// ---------------------------------------------------------------------------
__global__ void knot_prep(const float* __restrict__ freq,
                          const float* __restrict__ phase,
                          const float* __restrict__ regw,
                          const float* __restrict__ radii,
                          float* __restrict__ ws) {
    __shared__ float sF[128], sP[128], sR[128];       // [d*16+k]
    const int t = threadIdx.x;                        // 0..255
    if (t < 128) {
        const int k = t & 15, d = t >> 4;
        const float INV_2PI = 0.15915494309189535f;
        const float PHI_INV = 0.6180339887498949f;    // phi^-1
        float cf = 0.f, cp = 0.f, l = 1.f;
        for (int j = 0; j <= d; ++j) {
            cf = __builtin_fmaf(freq [k * D_SZ + j], l, cf);
            cp = __builtin_fmaf(phase[k * D_SZ + j], l, cp);
            l *= PHI_INV;
        }
        sF[t] = cf * INV_2PI;
        sP[t] = cp * INV_2PI;
        sR[t] = regw[d * K_SZ + k];                   // regWeights [D,K,1]
    }
    __syncthreads();

    // F'/P' tables in B-fragment per-lane order (128 entries each).
    if (t < 128) {
        const int m = t & 1, v = (t >> 1) & 7, c = (t >> 4) & 3, g = t >> 6;
        const int d = 2 * c + g;
        const int k = 2 * v + m;
        ws[t]       = sF[d * 16 + k];
        ws[128 + t] = sP[d * 16 + k];
    }

    // A fragments: 1024 half2-packed dwords.
    unsigned* wa = (unsigned*)(ws + 256);
    for (int q = 0; q < 4; ++q) {
        const int idx = t + q * 256;                  // 0..1023 = (c*8+v)*32+L
        const int L = idx & 31, v = (idx >> 5) & 7, c = idx >> 8;
        const int M = L & 15, g = L >> 4;
        const int K = 2 * v + 16 * g;
        const int d  = 2 * c + (K >> 4);              // K,K+1 share d (K even)
        const int k0 = K & 15;
        float h0 = (k0     == M) ? sR[d * 16 + M] : 0.f;
        float h1 = (k0 + 1 == M) ? sR[d * 16 + M] : 0.f;
        union { __fp16 h[2]; unsigned u; } pk;
        pk.h[0] = (__fp16)h0;
        pk.h[1] = (__fp16)h1;
        wa[idx] = pk.u;
    }

    if (t < K_SZ) ws[1280 + t] = radii[t];
}

// ---------------------------------------------------------------------------
// Main kernel: each wave32 owns 16 tiles of 16 samples within one batch.
// Per tile: 64 v_fma (pos) + 64 v_cos_f32 + 32 v_cvt_pkrtz + 4 v_wmma
// (f16 A/B, f32 accumulate), C seeded with knotRadii.
// ---------------------------------------------------------------------------
__global__ void __launch_bounds__(256)
knot_wmma(const float* __restrict__ x,
          const float* __restrict__ ws,
          float* __restrict__ out) {
    const int t    = threadIdx.x;
    const int lane = t & 31;
    const int g    = lane >> 4;               // half-lane group
    const int n    = lane & 15;               // sample column N
    const int gw   = blockIdx.x * 8 + (t >> 5);       // global wave 0..4095
    const int b    = gw >> 9;                          // 512 waves per batch
    const int sb   = (gw & 511) * 256;                 // 16 tiles * 16 samples

    // Per-lane F'/P' params (B-fragment order), loaded once, kept in VGPRs.
    float fA[4][8][2], pA[4][8][2];
#pragma unroll
    for (int c = 0; c < 4; ++c)
#pragma unroll
        for (int v = 0; v < 8; ++v) {
            const int idx = ((g * 4 + c) * 8 + v) * 2;
            fA[c][v][0] = ws[idx];       fA[c][v][1] = ws[idx + 1];
            pA[c][v][0] = ws[128 + idx]; pA[c][v][1] = ws[128 + idx + 1];
        }

    // A fragments (block-diagonal reg weights), loaded once.
    union AFrag { unsigned u[8]; v16h h; } a[4];
    const unsigned* wa = (const unsigned*)(ws + 256);
#pragma unroll
    for (int c = 0; c < 4; ++c)
#pragma unroll
        for (int v = 0; v < 8; ++v)
            a[c].u[v] = wa[(c * 8 + v) * 32 + lane];

    // C initializer = knotRadii in C-fragment order (M = r + 8g).
    float cinit[8];
#pragma unroll
    for (int r = 0; r < 8; ++r) cinit[r] = ws[1280 + 8 * g + r];

    const float* xb = x   + (size_t)b * S_SZ + sb;
    float*       ob = out + (size_t)b * (K_SZ * S_SZ) + sb;

    for (int tile = 0; tile < 16; ++tile) {
        const float xv = xb[tile * 16 + n];   // lanes 16-31 dup lanes 0-15
        v8f acc;
#pragma unroll
        for (int r = 0; r < 8; ++r) acc[r] = cinit[r];

#pragma unroll
        for (int c = 0; c < 4; ++c) {
            union BFrag { h2v h[8]; v16h v; } bf;
#pragma unroll
            for (int v = 0; v < 8; ++v) {
                // This lane's B rows: d = 2c+g, k' = 2v, 2v+1 (K = 16g+2v+m).
                const float c0 = __builtin_amdgcn_cosf(
                    __builtin_fmaf(xv, fA[c][v][0], pA[c][v][0]));
                const float c1 = __builtin_amdgcn_cosf(
                    __builtin_fmaf(xv, fA[c][v][1], pA[c][v][1]));
#if __has_builtin(__builtin_amdgcn_cvt_pkrtz)
                bf.h[v] = __builtin_amdgcn_cvt_pkrtz(c0, c1);
#else
                h2v tmp; tmp.x = (__fp16)c0; tmp.y = (__fp16)c1;
                bf.h[v] = tmp;
#endif
            }
            // D(16x16 f32) += A(16x32 f16, block-diag reg) x B(32x16 f16, cos)
            acc = __builtin_amdgcn_wmma_f32_16x16x32_f16(
                false, a[c].h, false, bf.v, (short)0, acc, false, false);
        }

        // Store: lane writes rows M = r + 8g, column s = sb + tile*16 + n.
#pragma unroll
        for (int r = 0; r < 8; ++r)
            __builtin_nontemporal_store(
                acc[r], ob + (size_t)(r + 8 * g) * S_SZ + tile * 16 + n);
    }
}

// ---------------------------------------------------------------------------
extern "C" void kernel_launch(void* const* d_in, const int* in_sizes, int n_in,
                              void* d_out, int out_size, void* d_ws, size_t ws_size,
                              hipStream_t stream) {
    (void)in_sizes; (void)n_in; (void)out_size; (void)ws_size;
    const float* x     = (const float*)d_in[0];
    const float* freq  = (const float*)d_in[1];
    const float* phase = (const float*)d_in[2];
    const float* regw  = (const float*)d_in[3];
    const float* radii = (const float*)d_in[4];
    float* out = (float*)d_out;
    float* ws  = (float*)d_ws;   // needs 1296 floats ~= 5.2 KB

    knot_prep<<<1, 256, 0, stream>>>(freq, phase, regw, radii, ws);

    // 512 blocks * 8 waves * 16 tiles * 16 samples = B*S = 1,048,576
    knot_wmma<<<512, 256, 0, stream>>>(x, ws, out);
}
